// SpikingBiPhysformerBlock_65506841198945
// MI455X (gfx1250) — matrix operand order repaired and checked by reading, not compiled, not run on hardware
//
#include <hip/hip_runtime.h>
#include <hip/hip_bf16.h>
#include <math.h>

// ---------------------------------------------------------------------------
// BiFormer block for MI455X (gfx1250, wave32, WMMA).
// All GEMM stages use v_wmma_f32_16x16x32_bf16 (bf16 in, f32 accum).
// GEMMs are fed by a double-buffered LDS pipeline using the CDNA5 async
// global->LDS copies (ASYNCcnt) when the toolchain exposes them.
// ---------------------------------------------------------------------------

typedef __bf16 bf16;
typedef __attribute__((ext_vector_type(16))) __bf16 bf16x16;
typedef __attribute__((ext_vector_type(8)))  __bf16 bf16x8;
typedef __attribute__((ext_vector_type(8)))  float  f32x8;
typedef __attribute__((ext_vector_type(4)))  int    v4i;

#define TOKENS   25088   // 8*56*56
#define TOKS_B   3136    // 56*56
#define CDIM     256
#define QKVC     768
#define NHEAD    8
#define HDIM     32
#define NREG     49
#define NBATCH   8

#if defined(__AMDGCN__) && __has_builtin(__builtin_amdgcn_global_load_async_to_lds_b128)
#define HAVE_ASYNC_LDS 1
#else
#define HAVE_ASYNC_LDS 0
#endif

#if HAVE_ASYNC_LDS
typedef __attribute__((address_space(1))) v4i gas_v4i;  // global
typedef __attribute__((address_space(3))) v4i las_v4i;  // LDS

// one instruction: each lane copies 16B global -> 16B LDS, tracked by ASYNCcnt
__device__ inline void async_b128(const bf16* g, bf16* l) {
  const v4i* gp = (const v4i*)g;
  v4i* lp = (v4i*)l;
  __builtin_amdgcn_global_load_async_to_lds_b128((gas_v4i*)gp, (las_v4i*)lp, 0, 0);
}
template <int N>
__device__ inline void wait_async() {
#if __has_builtin(__builtin_amdgcn_s_wait_asynccnt)
  __builtin_amdgcn_s_wait_asynccnt(N);
#else
  asm volatile("s_wait_asynccnt %0" ::"n"(N) : "memory");
#endif
}
#endif

// ---- WMMA fragment loaders (per CDNA5 ISA 7.12.2 VGPR layouts) -------------
// A 16x32 bf16: lane(half=lane>>4) holds K chunks [8h,8h+8) and [16+8h,+8)
__device__ inline bf16x16 frag_a(const bf16* p, int hlf) {
  union { bf16x16 v; bf16x8 h[2]; } u;
  u.h[0] = *(const bf16x8*)(p + 8 * hlf);
  u.h[1] = *(const bf16x8*)(p + 16 + 8 * hlf);
  return u.v;
}
// B 32x16 bf16 (column per lane): K chunks [8h,8h+8) and [16+8h,+8)
__device__ inline bf16x16 frag_b(const bf16* p, int hlf) {
  union { bf16x16 v; bf16x8 h[2]; } u;
  u.h[0] = *(const bf16x8*)(p + 8 * hlf);
  u.h[1] = *(const bf16x8*)(p + 16 + 8 * hlf);
  return u.v;
}

__device__ inline f32x8 wmma_bf16(bf16x16 a, bf16x16 b, f32x8 c) {
  return __builtin_amdgcn_wmma_f32_16x16x32_bf16(false, a, false, b, (short)0, c,
                                                 false, false);
}

__device__ inline float gelu_tanh(float x) {
  float x3 = x * x * x;
  return 0.5f * x * (1.f + tanhf(0.7978845608028654f * (x + 0.044715f * x3)));
}

// token row (within one batch image) of element rr of region R
__device__ inline int region_token(int R, int rr) {
  return ((R / 7) * 8 + (rr >> 3)) * 56 + (R % 7) * 8 + (rr & 7);
}

// ---- weight transpose + f32->bf16 :  w(K,N) -> wt(N,K) ---------------------
__global__ void wt_kernel(const float* __restrict__ w, bf16* __restrict__ wt,
                          int K, int N) {
  int i = blockIdx.x * 256 + threadIdx.x;
  if (i < K * N) {
    int k = i / N, n = i % N;
    wt[(size_t)n * K + k] = (bf16)w[i];
  }
}

// ---- LayerNorm over C=256, one block per token, bf16 out -------------------
__global__ void __launch_bounds__(256) ln_kernel(const float* __restrict__ x,
                                                 const float* __restrict__ g,
                                                 const float* __restrict__ bta,
                                                 bf16* __restrict__ out) {
  int tok = blockIdx.x, c = threadIdx.x;
  __shared__ float red[256];
  float v = x[(size_t)tok * CDIM + c];
  red[c] = v;
  __syncthreads();
  for (int s = 128; s > 0; s >>= 1) { if (c < s) red[c] += red[c + s]; __syncthreads(); }
  float mu = red[0] * (1.f / 256.f);
  __syncthreads();
  float dv = v - mu;
  red[c] = dv * dv;
  __syncthreads();
  for (int s = 128; s > 0; s >>= 1) { if (c < s) red[c] += red[c + s]; __syncthreads(); }
  float var = red[0] * (1.f / 256.f);
  out[(size_t)tok * CDIM + c] = (bf16)(dv * rsqrtf(var + 1e-5f) * g[c] + bta[c]);
}

// ---- generic bf16 WMMA GEMM: C[M,N] = A[M,K] * Bt[N,K]^T + epilogue --------
enum { EPI_BF16_BIAS = 0, EPI_F32_BIAS_RES = 1, EPI_BF16_GELU = 2 };

template <int MODE>
__global__ void __launch_bounds__(256) gemm_kernel(
    const bf16* __restrict__ A, const bf16* __restrict__ Bt,
    const float* __restrict__ bias, const float* __restrict__ res,
    void* __restrict__ outv, int M, int N, int K) {
  const int tid = threadIdx.x;
  const int lane = tid & 31, wave = tid >> 5;
  const int hlf = lane >> 4, l16 = lane & 15;
  const int wm = wave & 1, wn = wave >> 1;              // 2x4 wave grid
  const int rowBase = blockIdx.y * 128 + wm * 64;       // wave: 64 rows
  const int colBase = blockIdx.x * 128 + wn * 32;       // wave: 32 cols

  f32x8 acc[4][2];
  f32x8 z = {};
#pragma unroll
  for (int i = 0; i < 4; ++i)
#pragma unroll
    for (int j = 0; j < 2; ++j) acc[i][j] = z;

#if HAVE_ASYNC_LDS
  // double-buffered 128x32 tiles of A and Bt, streamed by async copies
  __shared__ __align__(16) bf16 sA[2][128][32];
  __shared__ __align__(16) bf16 sB[2][128][32];
  const int sr = tid >> 1;            // staged row (0..127)
  const int se = (tid & 1) * 16;      // staged element offset (0 or 16)
  const bf16* gA = A + (size_t)(blockIdx.y * 128 + sr) * K + se;
  const bf16* gB = Bt + (size_t)(blockIdx.x * 128 + sr) * K + se;

  const int nk = K >> 5;
  // prefetch tile 0 (4 async-load instructions per wave)
  async_b128(gA, &sA[0][sr][se]);
  async_b128(gA + 8, &sA[0][sr][se + 8]);
  async_b128(gB, &sB[0][sr][se]);
  async_b128(gB + 8, &sB[0][sr][se + 8]);

  for (int kk = 0; kk < nk; ++kk) {
    const int cur = kk & 1;
    if (kk + 1 < nk) {
      const int nxt = cur ^ 1;
      const int k1 = (kk + 1) << 5;
      async_b128(gA + k1, &sA[nxt][sr][se]);
      async_b128(gA + k1 + 8, &sA[nxt][sr][se + 8]);
      async_b128(gB + k1, &sB[nxt][sr][se]);
      async_b128(gB + k1 + 8, &sB[nxt][sr][se + 8]);
      wait_async<4>();   // oldest 4 (current tile) complete, next stays in flight
    } else {
      wait_async<0>();
    }
    __syncthreads();     // tile visible to all waves

    bf16x16 af[4], bfr[2];
#pragma unroll
    for (int i = 0; i < 4; ++i) af[i] = frag_a(&sA[cur][wm * 64 + i * 16 + l16][0], hlf);
#pragma unroll
    for (int j = 0; j < 2; ++j) bfr[j] = frag_b(&sB[cur][wn * 32 + j * 16 + l16][0], hlf);
#pragma unroll
    for (int i = 0; i < 4; ++i)
#pragma unroll
      for (int j = 0; j < 2; ++j) acc[i][j] = wmma_bf16(af[i], bfr[j], acc[i][j]);

    __syncthreads();     // everyone done reading before buffer is overwritten
  }
#else
  const bf16* arow[4];
  const bf16* brow[2];
#pragma unroll
  for (int i = 0; i < 4; ++i) arow[i] = A + (size_t)(rowBase + i * 16 + l16) * K;
#pragma unroll
  for (int j = 0; j < 2; ++j) brow[j] = Bt + (size_t)(colBase + j * 16 + l16) * K;

  for (int k0 = 0; k0 < K; k0 += 32) {
    bf16x16 af[4], bfr[2];
#pragma unroll
    for (int i = 0; i < 4; ++i) af[i] = frag_a(arow[i] + k0, hlf);
#pragma unroll
    for (int j = 0; j < 2; ++j) bfr[j] = frag_b(brow[j] + k0, hlf);
#pragma unroll
    for (int i = 0; i < 4; ++i)
#pragma unroll
      for (int j = 0; j < 2; ++j) acc[i][j] = wmma_bf16(af[i], bfr[j], acc[i][j]);
  }
#endif

#pragma unroll
  for (int j = 0; j < 2; ++j) {
    int col = colBase + j * 16 + l16;
    float bc = bias[col];
#pragma unroll
    for (int i = 0; i < 4; ++i) {
#pragma unroll
      for (int v = 0; v < 8; ++v) {
        int row = rowBase + i * 16 + 8 * hlf + v;
        float val = acc[i][j][v] + bc;
        if (MODE == EPI_BF16_GELU) val = gelu_tanh(val);
        if (MODE == EPI_F32_BIAS_RES) {
          val += res[(size_t)row * N + col];
          ((float*)outv)[(size_t)row * N + col] = val;
        } else {
          ((bf16*)outv)[(size_t)row * N + col] = (bf16)val;
        }
      }
    }
  }
}

// ---- region means of q and k (B*49 blocks x 256 threads) -------------------
__global__ void __launch_bounds__(256) regmean_kernel(const bf16* __restrict__ qkv,
                                                      float* __restrict__ qreg,
                                                      float* __restrict__ kreg) {
  int br = blockIdx.x;               // b*49 + R
  int b = br / NREG, R = br % NREG;
  int c = threadIdx.x;
  float sq = 0.f, sk = 0.f;
  for (int rr = 0; rr < 64; ++rr) {
    const bf16* p = qkv + (size_t)(b * TOKS_B + region_token(R, rr)) * QKVC;
    sq += (float)p[c];
    sk += (float)p[CDIM + c];
  }
  qreg[(size_t)br * CDIM + c] = sq * (1.f / 64.f);
  kreg[(size_t)br * CDIM + c] = sk * (1.f / 64.f);
}

// ---- routing logits + top-4 (B*49 blocks x 64 threads) ---------------------
__global__ void __launch_bounds__(64) route_kernel(const float* __restrict__ qreg,
                                                   const float* __restrict__ kreg,
                                                   int* __restrict__ topk) {
  int br = blockIdx.x;
  int b = br / NREG;
  __shared__ float sl[NREG];
  int s = threadIdx.x;
  if (s < NREG) {
    const float* qp = qreg + (size_t)br * CDIM;
    const float* kp = kreg + (size_t)(b * NREG + s) * CDIM;
    float a = 0.f;
    for (int c = 0; c < CDIM; ++c) a += qp[c] * kp[c];
    sl[s] = a;
  }
  __syncthreads();
  if (threadIdx.x == 0) {
    bool used[NREG];
    for (int i = 0; i < NREG; ++i) used[i] = false;
    for (int t = 0; t < 4; ++t) {
      int best = 0; float bv = -3.4e38f;
      for (int s2 = 0; s2 < NREG; ++s2)
        if (!used[s2] && sl[s2] > bv) { bv = sl[s2]; best = s2; }
      used[best] = true;
      topk[br * 4 + t] = best;
    }
  }
}

// ---- gathered attention: one block per (head, region, batch), 4 waves ------
__global__ void __launch_bounds__(128) attn_kernel(const bf16* __restrict__ qkv,
                                                   const int* __restrict__ topk,
                                                   bf16* __restrict__ xattn) {
  const int head = blockIdx.x, rq = blockIdx.y, b = blockIdx.z;
  const int tid = threadIdx.x;
  const int lane = tid & 31, wave = tid >> 5;      // wave = q M-tile (0..3)
  const int hlf = lane >> 4, l16 = lane & 15;

  __shared__ __align__(16) bf16 sAttn[64][256];    // softmaxed probs, bf16
  __shared__ __align__(16) bf16 sVT[HDIM][264];    // gathered V^T (+pad)
  __shared__ int sTok[256];                        // gathered token rows

  const int* myidx = topk + (b * NREG + rq) * 4;
  for (int t = tid; t < 256; t += 128) {
    int reg = myidx[t >> 6];
    sTok[t] = b * TOKS_B + region_token(reg, t & 63);
  }
  __syncthreads();

  // stage V^T into LDS so PV B-frags are contiguous ds reads
  for (int t = tid; t < 256; t += 128) {
    const bf16* vp = qkv + (size_t)sTok[t] * QKVC + 2 * CDIM + head * HDIM;
    for (int dd = 0; dd < HDIM; ++dd) sVT[dd][t] = vp[dd];
  }

  // q A-frag (16 rows of this wave's q tile; K = d = 32, single step)
  int qtok = b * TOKS_B + region_token(rq, wave * 16 + l16);
  bf16x16 aq = frag_a(qkv + (size_t)qtok * QKVC + head * HDIM, hlf);

  // scores: 16 N-tiles over the 256 gathered keys
  f32x8 acc[16];
  f32x8 z = {};
#pragma unroll
  for (int nt = 0; nt < 16; ++nt) {
    const bf16* kp = qkv + (size_t)sTok[nt * 16 + l16] * QKVC + CDIM + head * HDIM;
    acc[nt] = wmma_bf16(aq, frag_b(kp, hlf), z);
  }

  // softmax (rows live across the 16 lanes of a half-wave)
  const float scale = 0.17677669529663687f;  // 1/sqrt(32)
#pragma unroll
  for (int nt = 0; nt < 16; ++nt)
#pragma unroll
    for (int v = 0; v < 8; ++v) acc[nt][v] *= scale;

#pragma unroll
  for (int v = 0; v < 8; ++v) {
    float mx = -3.4e38f;
#pragma unroll
    for (int nt = 0; nt < 16; ++nt) mx = fmaxf(mx, acc[nt][v]);
    for (int m = 1; m < 16; m <<= 1) mx = fmaxf(mx, __shfl_xor(mx, m, 16));
    float sum = 0.f;
#pragma unroll
    for (int nt = 0; nt < 16; ++nt) {
      float e = __expf(acc[nt][v] - mx);
      acc[nt][v] = e;
      sum += e;
    }
    for (int m = 1; m < 16; m <<= 1) sum += __shfl_xor(sum, m, 16);
    float inv = 1.f / sum;
    int row = wave * 16 + 8 * hlf + v;
#pragma unroll
    for (int nt = 0; nt < 16; ++nt)
      sAttn[row][nt * 16 + l16] = (bf16)(acc[nt][v] * inv);
  }
  __syncthreads();

  // out = attn(64x256) @ V(256x32): K loop 8 steps, 2 d-tiles
  f32x8 o0 = z, o1 = z;
#pragma unroll
  for (int kk = 0; kk < 8; ++kk) {
    bf16x16 aa = frag_a(&sAttn[wave * 16 + l16][kk * 32], hlf);
    o0 = wmma_bf16(aa, frag_b(&sVT[l16][kk * 32], hlf), o0);
    o1 = wmma_bf16(aa, frag_b(&sVT[16 + l16][kk * 32], hlf), o1);
  }

#pragma unroll
  for (int v = 0; v < 8; ++v) {
    int tok = b * TOKS_B + region_token(rq, wave * 16 + 8 * hlf + v);
    bf16* op = xattn + (size_t)tok * CDIM + head * HDIM;
    op[l16] = (bf16)o0[v];
    op[16 + l16] = (bf16)o1[v];
  }
}

// ---------------------------------------------------------------------------
extern "C" void kernel_launch(void* const* d_in, const int* in_sizes, int n_in,
                              void* d_out, int out_size, void* d_ws, size_t ws_size,
                              hipStream_t stream) {
  (void)in_sizes; (void)n_in; (void)out_size; (void)ws_size;
  const float* x    = (const float*)d_in[0];
  const float* ln1g = (const float*)d_in[1];
  const float* ln1b = (const float*)d_in[2];
  const float* wqkv = (const float*)d_in[3];
  const float* bqkv = (const float*)d_in[4];
  const float* wo   = (const float*)d_in[5];
  const float* bo   = (const float*)d_in[6];
  const float* ln2g = (const float*)d_in[7];
  const float* ln2b = (const float*)d_in[8];
  const float* w1   = (const float*)d_in[9];
  const float* bm1  = (const float*)d_in[10];
  const float* w2   = (const float*)d_in[11];
  const float* bm2  = (const float*)d_in[12];

  char* ws = (char*)d_ws;
  size_t off = 0;
  auto take = [&](size_t bytes) -> char* {
    char* p = ws + off;
    off = (off + bytes + 255) & ~(size_t)255;
    return p;
  };
  bf16* xn    = (bf16*)take((size_t)TOKENS * CDIM * 2);
  bf16* qkv   = (bf16*)take((size_t)TOKENS * QKVC * 2);
  float* qreg = (float*)take((size_t)NBATCH * NREG * CDIM * 4);
  float* kreg = (float*)take((size_t)NBATCH * NREG * CDIM * 4);
  int*  tkidx = (int*)take((size_t)NBATCH * NREG * 4 * 4);
  bf16* xatt  = (bf16*)take((size_t)TOKENS * CDIM * 2);
  float* y1   = (float*)take((size_t)TOKENS * CDIM * 4);
  bf16* xn2   = (bf16*)take((size_t)TOKENS * CDIM * 2);
  bf16* hbuf  = (bf16*)take((size_t)TOKENS * 1024 * 2);
  bf16* wqkvT = (bf16*)take((size_t)QKVC * CDIM * 2);
  bf16* woT   = (bf16*)take((size_t)CDIM * CDIM * 2);
  bf16* w1T   = (bf16*)take((size_t)1024 * CDIM * 2);
  bf16* w2T   = (bf16*)take((size_t)CDIM * 1024 * 2);

  // weight transpose/convert (N,K) bf16
  wt_kernel<<<(CDIM * QKVC + 255) / 256, 256, 0, stream>>>(wqkv, wqkvT, CDIM, QKVC);
  wt_kernel<<<(CDIM * CDIM + 255) / 256, 256, 0, stream>>>(wo, woT, CDIM, CDIM);
  wt_kernel<<<(CDIM * 1024 + 255) / 256, 256, 0, stream>>>(w1, w1T, CDIM, 1024);
  wt_kernel<<<(1024 * CDIM + 255) / 256, 256, 0, stream>>>(w2, w2T, 1024, CDIM);

  // LN1 -> bf16
  ln_kernel<<<TOKENS, 256, 0, stream>>>(x, ln1g, ln1b, xn);

  // qkv = xn @ wqkv + bqkv  (bf16 out)
  gemm_kernel<EPI_BF16_BIAS><<<dim3(QKVC / 128, TOKENS / 128), 256, 0, stream>>>(
      xn, wqkvT, bqkv, nullptr, qkv, TOKENS, QKVC, CDIM);

  // routing
  regmean_kernel<<<NBATCH * NREG, 256, 0, stream>>>(qkv, qreg, kreg);
  route_kernel<<<NBATCH * NREG, 64, 0, stream>>>(qreg, kreg, tkidx);

  // gathered attention
  attn_kernel<<<dim3(NHEAD, NREG, NBATCH), 128, 0, stream>>>(qkv, tkidx, xatt);

  // y1 = x + attn @ wo + bo  (f32)
  gemm_kernel<EPI_F32_BIAS_RES><<<dim3(CDIM / 128, TOKENS / 128), 256, 0, stream>>>(
      xatt, woT, bo, x, y1, TOKENS, CDIM, CDIM);

  // LN2 -> bf16
  ln_kernel<<<TOKENS, 256, 0, stream>>>(y1, ln2g, ln2b, xn2);

  // h = gelu(xn2 @ w1 + bm1) (bf16)
  gemm_kernel<EPI_BF16_GELU><<<dim3(1024 / 128, TOKENS / 128), 256, 0, stream>>>(
      xn2, w1T, bm1, nullptr, hbuf, TOKENS, 1024, CDIM);

  // out = y1 + h @ w2 + bm2 (f32)
  gemm_kernel<EPI_F32_BIAS_RES><<<dim3(CDIM / 128, TOKENS / 128), 256, 0, stream>>>(
      hbuf, w2T, bm2, y1, (float*)d_out, TOKENS, CDIM, 1024);
}
